// Qwen3MoeSparseMoeBlock_61254823576068
// MI455X (gfx1250) — matrix operand order repaired and checked by reading, not compiled, not run on hardware
//
#include <hip/hip_runtime.h>
#include <hip/hip_bf16.h>
#include <math.h>

// ---------------------------------------------------------------------------
// Problem constants (from reference): B=1, S=2048 -> T=2048 tokens,
// H=2048 hidden, E=32 experts, I=768 intermediate, TOP_K=8.
// ---------------------------------------------------------------------------
#define T_TOK 2048
#define HDIM  2048
#define NEXP  32
#define IDIM  768
#define TOPK  8
#define MTILE 32        // tokens per expert block (two 16-row WMMA tiles)

typedef __attribute__((ext_vector_type(16))) __bf16 v16bf;
typedef __attribute__((ext_vector_type(8)))  __bf16 v8bf;
typedef __attribute__((ext_vector_type(4)))  __bf16 v4bf;
typedef __attribute__((ext_vector_type(8)))  float  v8f;
typedef __attribute__((ext_vector_type(16))) float  v16f;
typedef __attribute__((ext_vector_type(4)))  float  v4f;

// Convert 16 consecutive fp32 (64B, float4-aligned) to a v16bf fragment.
__device__ __forceinline__ v16bf cvt16_f32_bf16(const float* __restrict__ p) {
    v16f f = *(const v16f*)p;           // 4x global_load_b128
    v16bf r;
#pragma unroll
    for (int j = 0; j < 16; ++j) r[j] = (__bf16)f[j];   // v_cvt_pk_bf16_f32 x8
    return r;
}

__device__ __forceinline__ v16bf join8(v8bf lo, v8bf hi) {
    return __builtin_shufflevector(lo, hi, 0, 1, 2, 3, 4, 5, 6, 7,
                                   8, 9, 10, 11, 12, 13, 14, 15);
}

// ---------------------------------------------------------------------------
// Kernel 0: zero the output accumulator and the per-expert counters.
// ---------------------------------------------------------------------------
__global__ void moe_zero_kernel(float* __restrict__ out, int* __restrict__ counts) {
    const size_t tid = (size_t)blockIdx.x * blockDim.x + threadIdx.x;
    v4f z = {};
    *(v4f*)(out + tid * 4) = z;          // grid sized exactly: T*H/4 threads
    if (blockIdx.x == 0 && threadIdx.x < NEXP) counts[threadIdx.x] = 0;
}

// ---------------------------------------------------------------------------
// Kernel 1: router. One block (256 threads) per token.
// logits = x @ gate_w^T ; top-8 ; softmax over the selected 8 ;
// append (token, weight) to each selected expert's list.
// ---------------------------------------------------------------------------
__global__ void moe_router_kernel(const float* __restrict__ x,
                                  const float* __restrict__ gate_w,
                                  int*   __restrict__ counts,
                                  int*   __restrict__ ids,
                                  float* __restrict__ wts) {
    __shared__ float xr[HDIM];
    __shared__ float part[256];
    __shared__ float logits[NEXP];

    const int t   = blockIdx.x;
    const int tid = threadIdx.x;

    for (int h = tid; h < HDIM; h += 256) xr[h] = x[(size_t)t * HDIM + h];
    __syncthreads();

    // 8 threads per expert, each reduces a contiguous 256-chunk.
    const int e     = tid >> 3;
    const int lane8 = tid & 7;
    {
        const float* gw = gate_w + (size_t)e * HDIM + lane8 * 256;
        const float* xc = xr + lane8 * 256;
        float s = 0.0f;
#pragma unroll 8
        for (int h = 0; h < 256; ++h) s += xc[h] * gw[h];
        part[tid] = s;
    }
    __syncthreads();
    if (lane8 == 0) {
        float l = 0.0f;
#pragma unroll
        for (int j = 0; j < 8; ++j) l += part[tid + j];
        logits[e] = l;
    }
    __syncthreads();

    if (tid == 0) {
        float lg[NEXP];
#pragma unroll
        for (int j = 0; j < NEXP; ++j) lg[j] = logits[j];
        int   se[TOPK];
        float sw[TOPK];
#pragma unroll
        for (int k = 0; k < TOPK; ++k) {            // serial top-8 (32 entries)
            int   bi = 0;
            float bv = lg[0];
            for (int j = 1; j < NEXP; ++j)
                if (lg[j] > bv) { bv = lg[j]; bi = j; }
            se[k] = bi; sw[k] = bv; lg[bi] = -INFINITY;
        }
        const float m = sw[0];
        float denom = 0.0f;
#pragma unroll
        for (int k = 0; k < TOPK; ++k) { sw[k] = __expf(sw[k] - m); denom += sw[k]; }
        const float inv = __builtin_amdgcn_rcpf(denom);
#pragma unroll
        for (int k = 0; k < TOPK; ++k) {
            const int ex  = se[k];
            const int pos = atomicAdd(&counts[ex], 1);
            ids[ex * T_TOK + pos] = t;
            wts[ex * T_TOK + pos] = sw[k] * inv;
        }
    }
}

// ---------------------------------------------------------------------------
// Kernel 2: fused expert GLU MLP via bf16 WMMA.
// One block per (expert, 32-token tile); 256 threads = 8 waves.
// Each wave computes 32xN output tiles as two 16x16 WMMA tiles sharing B,
// halving fp32->bf16 weight-conversion VALU per WMMA and per-expert
// weight re-reads.
//
// Dynamic LDS layout (~176 KB, fits 320 KB/WGP):
//   [0,128)        int   toks[32]
//   [128,256)      float twt[32]
//   [256,+128K)    bf16  xs[32][2048]    gathered token rows
//   [+48K)         bf16  act[32][768]    silu(g)*u*w
// ---------------------------------------------------------------------------
#define SMEM_TOKS  0
#define SMEM_TWT   128
#define SMEM_XS    256
#define SMEM_ACT   (256 + MTILE * HDIM * 2)
#define SMEM_TOTAL (SMEM_ACT + MTILE * IDIM * 2)

__global__ void __launch_bounds__(256, 1)
moe_expert_kernel(const float* __restrict__ x,
                  const float* __restrict__ w_gate,
                  const float* __restrict__ w_up,
                  const float* __restrict__ w_down,
                  const int*   __restrict__ counts,
                  const int*   __restrict__ ids,
                  const float* __restrict__ wts,
                  float*       __restrict__ out) {
    extern __shared__ char smem[];
    int*    toks = (int*)(smem + SMEM_TOKS);
    float*  twt  = (float*)(smem + SMEM_TWT);
    __bf16* xs   = (__bf16*)(smem + SMEM_XS);
    __bf16* act  = (__bf16*)(smem + SMEM_ACT);

    const int e     = blockIdx.y;
    const int cnt   = counts[e];
    const int start = blockIdx.x * MTILE;
    if (start >= cnt) return;                    // uniform over block: safe
    const int nv  = min(MTILE, cnt - start);
    const int tid = threadIdx.x;

    if (tid < MTILE) {
        const int src = (tid < nv) ? (start + tid) : start;   // pad w/ 0-wgt
        toks[tid] = ids[e * T_TOK + src];
        twt[tid]  = (tid < nv) ? wts[e * T_TOK + src] : 0.0f;
    }
    __syncthreads();

    // ---- Gather 32 token rows, fp32 -> bf16, into LDS (vectorized) ----
    for (int idx = tid; idx < MTILE * (HDIM / 4); idx += 256) {
        const int row  = idx >> 9;               // /512
        const int col4 = idx & 511;
        v4f f = *(const v4f*)(x + (size_t)toks[row] * HDIM + col4 * 4);
        v4bf b;
#pragma unroll
        for (int j = 0; j < 4; ++j) b[j] = (__bf16)f[j];
        *(v4bf*)(xs + row * HDIM + col4 * 4) = b;
    }
    __syncthreads();

    const int wave = tid >> 5;
    const int L    = tid & 31;
    const int n16  = L & 15;                     // N (col) / M (row) index
    const int grpA = (L < 16) ? 0 : 16;          // A-fragment K split
    const int grpB = (L < 16) ? 0 : 16;          // B-fragment K split
    const int moff = (L < 16) ? 0 : 8;           // D-fragment row offset

    // =============== Phase 1: act = silu(x@Wg^T) * (x@Wu^T) * w ===========
    for (int nt = wave; nt < IDIM / 16; nt += 8) {
        const int n_base = nt * 16;
        const float* gq = w_gate + ((size_t)e * IDIM + n_base + n16) * HDIM;
        const float* uq = w_up   + ((size_t)e * IDIM + n_base + n16) * HDIM;
        v8f cg0 = {}, cg1 = {}, cu0 = {}, cu1 = {};
        for (int kc = 0; kc < HDIM / 32; ++kc) {
            const int k0 = kc * 32;
            // Two A fragments (rows 0-15 and 16-31) from LDS
            const __bf16* xr0 = xs + n16 * HDIM + k0 + grpA;
            const __bf16* xr1 = xs + (16 + n16) * HDIM + k0 + grpA;
            v16bf a0 = join8(*(const v8bf*)xr0, *(const v8bf*)(xr0 + 8));
            v16bf a1 = join8(*(const v8bf*)xr1, *(const v8bf*)(xr1 + 8));
            // One converted B pair shared by both A tiles (4 WMMAs / 16 cvt)
            v16bf bg = cvt16_f32_bf16(gq + k0 + grpB);
            v16bf bu = cvt16_f32_bf16(uq + k0 + grpB);
            cg0 = __builtin_amdgcn_wmma_f32_16x16x32_bf16(
                      false, a0, false, bg, (short)0, cg0, false, false);
            cg1 = __builtin_amdgcn_wmma_f32_16x16x32_bf16(
                      false, a1, false, bg, (short)0, cg1, false, false);
            cu0 = __builtin_amdgcn_wmma_f32_16x16x32_bf16(
                      false, a0, false, bu, (short)0, cu0, false, false);
            cu1 = __builtin_amdgcn_wmma_f32_16x16x32_bf16(
                      false, a1, false, bu, (short)0, cu1, false, false);
        }
        // Epilogue: silu(g)*u*route_weight -> bf16 act in LDS (fast rcp)
#pragma unroll
        for (int v = 0; v < 8; ++v) {
            {
                const int m = v + moff;
                const float g = cg0[v];
                const float s = g * __builtin_amdgcn_rcpf(1.0f + __expf(-g));
                act[m * IDIM + n_base + n16] = (__bf16)(s * cu0[v] * twt[m]);
            }
            {
                const int m = 16 + v + moff;
                const float g = cg1[v];
                const float s = g * __builtin_amdgcn_rcpf(1.0f + __expf(-g));
                act[m * IDIM + n_base + n16] = (__bf16)(s * cu1[v] * twt[m]);
            }
        }
    }
    __syncthreads();

    // =============== Phase 2: out[t,:] += act @ Wd^T (atomic scatter) =====
    for (int nt = wave; nt < HDIM / 16; nt += 8) {
        const int n_base = nt * 16;
        const float* dq = w_down + ((size_t)e * HDIM + n_base + n16) * IDIM;
        v8f c0 = {}, c1 = {};
        for (int kc = 0; kc < IDIM / 32; ++kc) {
            const int k0 = kc * 32;
            const __bf16* ar0 = act + n16 * IDIM + k0 + grpA;
            const __bf16* ar1 = act + (16 + n16) * IDIM + k0 + grpA;
            v16bf a0 = join8(*(const v8bf*)ar0, *(const v8bf*)(ar0 + 8));
            v16bf a1 = join8(*(const v8bf*)ar1, *(const v8bf*)(ar1 + 8));
            v16bf b  = cvt16_f32_bf16(dq + k0 + grpB);
            c0 = __builtin_amdgcn_wmma_f32_16x16x32_bf16(
                     false, a0, false, b, (short)0, c0, false, false);
            c1 = __builtin_amdgcn_wmma_f32_16x16x32_bf16(
                     false, a1, false, b, (short)0, c1, false, false);
        }
#pragma unroll
        for (int v = 0; v < 8; ++v) {
            const int m0 = v + moff;
            const int m1 = 16 + v + moff;
            if (m0 < nv)
                atomicAdd(&out[(size_t)toks[m0] * HDIM + n_base + n16], c0[v]);
            if (m1 < nv)
                atomicAdd(&out[(size_t)toks[m1] * HDIM + n_base + n16], c1[v]);
        }
    }
}

// ---------------------------------------------------------------------------
// Host launcher
// ---------------------------------------------------------------------------
extern "C" void kernel_launch(void* const* d_in, const int* in_sizes, int n_in,
                              void* d_out, int out_size, void* d_ws, size_t ws_size,
                              hipStream_t stream) {
    const float* x       = (const float*)d_in[0];   // [1,2048,2048]
    const float* gate_w  = (const float*)d_in[1];   // [32,2048]
    const float* w_gate  = (const float*)d_in[2];   // [32,768,2048]
    const float* w_up    = (const float*)d_in[3];   // [32,768,2048]
    const float* w_down  = (const float*)d_in[4];   // [32,2048,768]
    float*       out     = (float*)d_out;           // [1,2048,2048]

    // Workspace layout
    char* ws = (char*)d_ws;
    int*   counts = (int*)ws;                                   // 32 ints
    int*   ids    = (int*)(ws + 128);                           // 32*2048 ints
    float* wts    = (float*)(ws + 128 + NEXP * T_TOK * 4);      // 32*2048 f32

    // 0) zero output accumulator + counters
    moe_zero_kernel<<<dim3((T_TOK * HDIM) / (256 * 4)), dim3(256), 0, stream>>>(
        out, counts);

    // 1) routing: top-8 softmax -> per-expert token lists
    moe_router_kernel<<<dim3(T_TOK), dim3(256), 0, stream>>>(
        x, gate_w, counts, ids, wts);

    // 2) fused expert GLU MLP (bf16 WMMA), expert-major grid for L2 reuse
    moe_expert_kernel<<<dim3(T_TOK / MTILE, NEXP), dim3(256), SMEM_TOTAL, stream>>>(
        x, w_gate, w_up, w_down, counts, ids, wts, out);
}
